// MiniBatchDiscrimination_15917148799715
// MI455X (gfx1250) — compile-verified, matrix operand order checked by reading, not run on hardware
//
#include <hip/hip_runtime.h>

#define N_PTS 8192
#define D_DIM 128
#define BT    128          // block tile (M and N)
#define LSTR  136          // padded LDS row stride in elements (272B -> conflict-free)
#define TILE_ELEMS (BT * LSTR)
#define SMEM_BYTES (4 * TILE_ELEMS * 2)

typedef __attribute__((ext_vector_type(16))) __bf16        v16bf;
typedef __attribute__((ext_vector_type(8)))  float          v8f;
typedef __attribute__((ext_vector_type(4))) unsigned int    u32x4;
typedef __attribute__((ext_vector_type(8)))  int            i32x8;
typedef __attribute__((ext_vector_type(4)))  int            i32x4;
typedef __attribute__((ext_vector_type(4))) unsigned short  ushort4v;

union FragU { u32x4 u[2]; v16bf v; };

__device__ __forceinline__ unsigned short f2bf(float f) {
    unsigned int u = __float_as_uint(f);
    u += 0x7FFFu + ((u >> 16) & 1u);                 // round-to-nearest-even
    return (unsigned short)(u >> 16);
}
__device__ __forceinline__ float bf2f(unsigned short h) {
    return __uint_as_float(((unsigned int)h) << 16);
}

// ---- TDM: load one 128x128 bf16 tile (row-major, 128-elem rows) into LDS ----
// LDS gets 16B padding after each 256B row => effective row stride 136 elems.
__device__ __forceinline__ void tdm_load_tile(const unsigned short* gtile,
                                              unsigned lds_off) {
    const unsigned long long ga = (unsigned long long)(uintptr_t)gtile;
    u32x4 g0;
    g0[0] = 1u;                                      // count=1, user mode
    g0[1] = lds_off;                                 // lds_addr (bytes)
    g0[2] = (unsigned)(ga & 0xFFFFFFFFu);            // global_addr[31:0]
    g0[3] = (unsigned)((ga >> 32) & 0x01FFFFFFu)     // global_addr[56:32]
            | (2u << 30);                            // type = 2 ("image")
    i32x8 g1;
    g1[0] = (1 << 16)                                // data_size = 2 bytes
          | (1 << 20)                                // pad_enable
          | (5 << 22)                                // pad_interval: 64 DWORDs (256B)
          | (3 << 25);                               // pad_amount: 4 DWORDs (16B)
    g1[1] = (int)(128u << 16);                       // tensor_dim0 = 128 (lo16)
    g1[2] = (int)(8192u << 16);                      // dim0 hi=0 | tensor_dim1 lo16
    g1[3] = (int)(128u << 16);                       // dim1 hi=0 | tile_dim0 = 128
    g1[4] = 128;                                     // tile_dim1 = 128, tile_dim2 = 0
    g1[5] = 128;                                     // tensor_dim0_stride = 128
    g1[6] = 0;                                       // stride0 hi | stride1 lo
    g1[7] = 0;
    const i32x4 z4 = {0, 0, 0, 0};                   // groups 2/3 unused (2D)
    const i32x8 z8 = {0, 0, 0, 0, 0, 0, 0, 0};
    __builtin_amdgcn_tensor_load_to_lds(g0, g1, z4, z4, z8, 0);
}

// ---------------- kernel 1: row norms + split-bf16 precompute ----------------
__global__ __launch_bounds__(256) void prep_kernel(
        const float* __restrict__ x, float* __restrict__ sq,
        unsigned short* __restrict__ hi, unsigned short* __restrict__ lo) {
    const int row  = blockIdx.x * 8 + (threadIdx.x >> 5);
    const int lane = threadIdx.x & 31;
    const float4 v = *(const float4*)(x + row * D_DIM + lane * 4);
    const float f[4] = {v.x, v.y, v.z, v.w};
    ushort4v h4, l4;
    float s = 0.0f;
    #pragma unroll
    for (int j = 0; j < 4; ++j) {
        s += f[j] * f[j];
        const unsigned short h = f2bf(f[j]);
        h4[j] = h;
        l4[j] = f2bf(f[j] - bf2f(h));
    }
    *(ushort4v*)(hi + row * D_DIM + lane * 4) = h4;
    *(ushort4v*)(lo + row * D_DIM + lane * 4) = l4;
    #pragma unroll
    for (int off = 16; off > 0; off >>= 1) s += __shfl_xor(s, off);
    if (lane == 0) sq[row] = s;
}

// -------- kernel 2 (fast path): TDM staging + split-bf16 WMMA + exp ---------
__global__ __launch_bounds__(256) void pair_exp_tdm_kernel(
        const unsigned short* __restrict__ hi, const unsigned short* __restrict__ lo,
        const float* __restrict__ sq, float* __restrict__ partials) {
    extern __shared__ unsigned short smem[];
    __shared__ float red[256];

    unsigned short* shAhi = smem;
    unsigned short* shAlo = smem + TILE_ELEMS;
    unsigned short* shBhi = smem + 2 * TILE_ELEMS;
    unsigned short* shBlo = smem + 3 * TILE_ELEMS;

    const int tid   = threadIdx.x;
    const int lane  = tid & 31;
    const int wave  = tid >> 5;
    const int wm    = wave & 3;                 // 4 M-groups of 32 rows
    const int wn    = wave >> 2;                // 2 N-groups of 64 cols
    const int rowA0 = blockIdx.y * BT;
    const int rowB0 = blockIdx.x * BT;

    // ---- stage all four full-K tiles via the Tensor Data Mover ----
    if (wave == 0) {
        const unsigned base = (unsigned)(uintptr_t)smem;
        tdm_load_tile(hi + rowA0 * D_DIM, base);
        tdm_load_tile(lo + rowA0 * D_DIM, base + 1u * TILE_ELEMS * 2u);
        tdm_load_tile(hi + rowB0 * D_DIM, base + 2u * TILE_ELEMS * 2u);
        tdm_load_tile(lo + rowB0 * D_DIM, base + 3u * TILE_ELEMS * 2u);
        __builtin_amdgcn_s_wait_tensorcnt(0);
    }
    __syncthreads();

    const v8f vzero = {};
    v8f acc[2][4];
    #pragma unroll
    for (int mi = 0; mi < 2; ++mi)
        #pragma unroll
        for (int ni = 0; ni < 4; ++ni) acc[mi][ni] = vzero;

    const int rsel  = lane & 15;                // row/col within 16x16 fragment
    const int khalf = (lane >> 4) << 3;         // K-half select per ISA layout

    #pragma unroll
    for (int ks = 0; ks < D_DIM / 32; ++ks) {   // pure LDS + WMMA, no barriers
        const int k0 = ks * 32;
        v16bf ahi[2], alo[2], bhi[4], blo[4];
        #pragma unroll
        for (int mi = 0; mi < 2; ++mi) {
            const int r = wm * 32 + mi * 16 + rsel;
            const unsigned short* p = shAhi + r * LSTR + k0 + khalf;
            const unsigned short* q = shAlo + r * LSTR + k0 + khalf;
            FragU fh, fl;
            fh.u[0] = *(const u32x4*)p;  fh.u[1] = *(const u32x4*)(p + 16);
            fl.u[0] = *(const u32x4*)q;  fl.u[1] = *(const u32x4*)(q + 16);
            ahi[mi] = fh.v;  alo[mi] = fl.v;
        }
        #pragma unroll
        for (int ni = 0; ni < 4; ++ni) {
            const int r = wn * 64 + ni * 16 + rsel;
            const unsigned short* p = shBhi + r * LSTR + k0 + khalf;
            const unsigned short* q = shBlo + r * LSTR + k0 + khalf;
            FragU fh, fl;
            fh.u[0] = *(const u32x4*)p;  fh.u[1] = *(const u32x4*)(p + 16);
            fl.u[0] = *(const u32x4*)q;  fl.u[1] = *(const u32x4*)(q + 16);
            bhi[ni] = fh.v;  blo[ni] = fl.v;
        }
        #pragma unroll
        for (int mi = 0; mi < 2; ++mi)
            #pragma unroll
            for (int ni = 0; ni < 4; ++ni) {
                acc[mi][ni] = __builtin_amdgcn_wmma_f32_16x16x32_bf16(
                    false, ahi[mi], false, bhi[ni], (short)0, acc[mi][ni],
                    false, false);
                acc[mi][ni] = __builtin_amdgcn_wmma_f32_16x16x32_bf16(
                    false, ahi[mi], false, blo[ni], (short)0, acc[mi][ni],
                    false, false);
                acc[mi][ni] = __builtin_amdgcn_wmma_f32_16x16x32_bf16(
                    false, alo[mi], false, bhi[ni], (short)0, acc[mi][ni],
                    false, false);
            }
    }

    // ---- epilogue: d = sq_i + sq_j - 2*dot ; sum exp(-max(d,0)) ----
    float local = 0.0f;
    const int mh = (lane >> 4) << 3;            // C layout: M = r + 8*(lane>>4)
    #pragma unroll
    for (int mi = 0; mi < 2; ++mi) {
        float sqM[8];
        #pragma unroll
        for (int r = 0; r < 8; ++r)
            sqM[r] = sq[rowA0 + wm * 32 + mi * 16 + mh + r];
        #pragma unroll
        for (int ni = 0; ni < 4; ++ni) {
            const float sqN = sq[rowB0 + wn * 64 + ni * 16 + rsel];
            #pragma unroll
            for (int r = 0; r < 8; ++r) {
                float d = sqM[r] + sqN - 2.0f * acc[mi][ni][r];
                d = fmaxf(d, 0.0f);
                local += __expf(-d);
            }
        }
    }

    red[tid] = local;
    __syncthreads();
    #pragma unroll
    for (int s = 128; s > 0; s >>= 1) {
        if (tid < s) red[tid] += red[tid + s];
        __syncthreads();
    }
    if (tid == 0) partials[blockIdx.y * 64 + blockIdx.x] = red[0];
}

// ------- kernel 2 (fallback, tiny ws): convert in-kernel (round-1 path) -----
__global__ __launch_bounds__(256) void sq_kernel(const float* __restrict__ x,
                                                 float* __restrict__ sq) {
    const int row  = blockIdx.x * 8 + (threadIdx.x >> 5);
    const int lane = threadIdx.x & 31;
    const float4 v = *(const float4*)(x + row * D_DIM + lane * 4);
    float s = v.x * v.x + v.y * v.y + v.z * v.z + v.w * v.w;
    #pragma unroll
    for (int off = 16; off > 0; off >>= 1) s += __shfl_xor(s, off);
    if (lane == 0) sq[row] = s;
}

#define FSTR 40
__global__ __launch_bounds__(256) void pair_exp_fallback_kernel(
        const float* __restrict__ x, const float* __restrict__ sq,
        float* __restrict__ partials) {
    __shared__ unsigned short shAhi[BT * FSTR];
    __shared__ unsigned short shAlo[BT * FSTR];
    __shared__ unsigned short shBhi[BT * FSTR];
    __shared__ unsigned short shBlo[BT * FSTR];
    __shared__ float red[256];

    const int tid = threadIdx.x, lane = tid & 31, wave = tid >> 5;
    const int wm = wave & 3, wn = wave >> 2;
    const int rowA0 = blockIdx.y * BT, rowB0 = blockIdx.x * BT;
    const v8f vzero = {};
    v8f acc[2][4];
    #pragma unroll
    for (int mi = 0; mi < 2; ++mi)
        #pragma unroll
        for (int ni = 0; ni < 4; ++ni) acc[mi][ni] = vzero;
    const int rsel = lane & 15, khalf = (lane >> 4) << 3;

    for (int ks = 0; ks < D_DIM / 32; ++ks) {
        if (ks) __syncthreads();
        const int kbase = ks * 32;
        #pragma unroll
        for (int t = 0; t < 2; ++t) {
            const int rb = t ? rowB0 : rowA0;
            unsigned short* dhi = t ? shBhi : shAhi;
            unsigned short* dlo = t ? shBlo : shAlo;
            for (int c = tid; c < (BT * 32) / 4; c += 256) {
                const int row = c >> 3, col = (c & 7) << 2;
                const float4 v = *(const float4*)(x + (rb + row) * D_DIM + kbase + col);
                const float f[4] = {v.x, v.y, v.z, v.w};
                ushort4v h4, l4;
                #pragma unroll
                for (int j = 0; j < 4; ++j) {
                    const unsigned short h = f2bf(f[j]);
                    h4[j] = h; l4[j] = f2bf(f[j] - bf2f(h));
                }
                const int a = row * FSTR + col;
                *(ushort4v*)(dhi + a) = h4;
                *(ushort4v*)(dlo + a) = l4;
            }
        }
        __syncthreads();
        v16bf ahi[2], alo[2], bhi[4], blo[4];
        #pragma unroll
        for (int mi = 0; mi < 2; ++mi) {
            const int r = wm * 32 + mi * 16 + rsel;
            const unsigned short* p = shAhi + r * FSTR + khalf;
            const unsigned short* q = shAlo + r * FSTR + khalf;
            FragU fh, fl;
            fh.u[0] = *(const u32x4*)p;  fh.u[1] = *(const u32x4*)(p + 16);
            fl.u[0] = *(const u32x4*)q;  fl.u[1] = *(const u32x4*)(q + 16);
            ahi[mi] = fh.v;  alo[mi] = fl.v;
        }
        #pragma unroll
        for (int ni = 0; ni < 4; ++ni) {
            const int r = wn * 64 + ni * 16 + rsel;
            const unsigned short* p = shBhi + r * FSTR + khalf;
            const unsigned short* q = shBlo + r * FSTR + khalf;
            FragU fh, fl;
            fh.u[0] = *(const u32x4*)p;  fh.u[1] = *(const u32x4*)(p + 16);
            fl.u[0] = *(const u32x4*)q;  fl.u[1] = *(const u32x4*)(q + 16);
            bhi[ni] = fh.v;  blo[ni] = fl.v;
        }
        #pragma unroll
        for (int mi = 0; mi < 2; ++mi)
            #pragma unroll
            for (int ni = 0; ni < 4; ++ni) {
                acc[mi][ni] = __builtin_amdgcn_wmma_f32_16x16x32_bf16(
                    false, ahi[mi], false, bhi[ni], (short)0, acc[mi][ni], false, false);
                acc[mi][ni] = __builtin_amdgcn_wmma_f32_16x16x32_bf16(
                    false, ahi[mi], false, blo[ni], (short)0, acc[mi][ni], false, false);
                acc[mi][ni] = __builtin_amdgcn_wmma_f32_16x16x32_bf16(
                    false, alo[mi], false, bhi[ni], (short)0, acc[mi][ni], false, false);
            }
    }

    float local = 0.0f;
    const int mh = (lane >> 4) << 3;
    #pragma unroll
    for (int mi = 0; mi < 2; ++mi) {
        float sqM[8];
        #pragma unroll
        for (int r = 0; r < 8; ++r)
            sqM[r] = sq[rowA0 + wm * 32 + mi * 16 + mh + r];
        #pragma unroll
        for (int ni = 0; ni < 4; ++ni) {
            const float sqN = sq[rowB0 + wn * 64 + ni * 16 + rsel];
            #pragma unroll
            for (int r = 0; r < 8; ++r) {
                float d = sqM[r] + sqN - 2.0f * acc[mi][ni][r];
                local += __expf(-fmaxf(d, 0.0f));
            }
        }
    }
    red[tid] = local;
    __syncthreads();
    #pragma unroll
    for (int s = 128; s > 0; s >>= 1) {
        if (tid < s) red[tid] += red[tid + s];
        __syncthreads();
    }
    if (tid == 0) partials[blockIdx.y * 64 + blockIdx.x] = red[0];
}

// ---------------- kernel 3: final deterministic reduce ----------------------
__global__ __launch_bounds__(256) void reduce_kernel(
        const float* __restrict__ partials, float* __restrict__ out) {
    __shared__ float red[256];
    float s = 0.0f;
    for (int i = threadIdx.x; i < 4096; i += 256) s += partials[i];
    red[threadIdx.x] = s;
    __syncthreads();
    #pragma unroll
    for (int st = 128; st > 0; st >>= 1) {
        if (threadIdx.x < st) red[threadIdx.x] += red[threadIdx.x + st];
        __syncthreads();
    }
    if (threadIdx.x == 0)
        out[0] = red[0] * (1.0f / ((float)N_PTS * (float)N_PTS));
}

extern "C" void kernel_launch(void* const* d_in, const int* in_sizes, int n_in,
                              void* d_out, int out_size, void* d_ws,
                              size_t ws_size, hipStream_t stream) {
    (void)in_sizes; (void)n_in; (void)out_size;
    const float* x  = (const float*)d_in[0];       // [8192,128] f32 (W unused)
    float* ws       = (float*)d_ws;
    float* sq       = ws;                          // 8192 f32
    float* partials = ws + N_PTS;                  // 4096 f32
    unsigned short* hi = (unsigned short*)(ws + N_PTS + 4096);
    unsigned short* lo = hi + N_PTS * D_DIM;

    const size_t need = (size_t)(N_PTS + 4096) * 4 + (size_t)2 * N_PTS * D_DIM * 2;
    dim3 grid(N_PTS / BT, N_PTS / BT);
    if (ws_size >= need) {
        prep_kernel<<<N_PTS / 8, 256, 0, stream>>>(x, sq, hi, lo);
        pair_exp_tdm_kernel<<<grid, 256, SMEM_BYTES, stream>>>(hi, lo, sq, partials);
    } else {
        sq_kernel<<<N_PTS / 8, 256, 0, stream>>>(x, sq);
        pair_exp_fallback_kernel<<<grid, 256, 0, stream>>>(x, sq, partials);
    }
    reduce_kernel<<<1, 256, 0, stream>>>(partials, (float*)d_out);
}
